// Unified_15040975470626
// MI455X (gfx1250) — compile-verified, matrix operand order checked by reading, not run on hardware
//
#include <hip/hip_runtime.h>
#include <math.h>

// ---------------------------------------------------------------------------
// Model constants (from reference)
// ---------------------------------------------------------------------------
#define BT   4096      // B*T
#define TT   2048      // T
#define EE   1024      // E
#define NIN  4112      // 4E + NE
#define NEXP 16
#define NH   8
#define DD   128
#define SS   256

typedef __attribute__((ext_vector_type(16))) __bf16          v16bf;
typedef __attribute__((ext_vector_type(16))) unsigned short  v16us;
typedef __attribute__((ext_vector_type(8)))  float           v8f;
typedef __attribute__((ext_vector_type(4)))  int             v4i;

#define AS1 __attribute__((address_space(1)))
#define AS3 __attribute__((address_space(3)))

#if __has_builtin(__builtin_amdgcn_global_load_async_to_lds_b128)
#define ASYNC_CP 1
#else
#define ASYNC_CP 0
#endif

// Async (or fallback synchronous) 16-byte global -> LDS copy.
// Generic LDS addresses carry the LDS byte offset in their low 32 bits
// (aperture rule), and addrspace(1) pointers are the plain 64-bit VA.
__device__ __forceinline__ void cp_async_b128(const unsigned short* gsrc,
                                              unsigned short* ldst) {
#if ASYNC_CP
    AS1 v4i* g = (AS1 v4i*)(unsigned long long)(size_t)gsrc;
    AS3 v4i* l = (AS3 v4i*)(unsigned)(size_t)ldst;
    __builtin_amdgcn_global_load_async_to_lds_b128(g, l, 0, 0);
#else
    *(uint4*)ldst = *(const uint4*)gsrc;
#endif
}

__device__ __forceinline__ void cp_async_wait() {
#if ASYNC_CP
#if __has_builtin(__builtin_amdgcn_s_wait_asynccnt)
    __builtin_amdgcn_s_wait_asynccnt(0);
#else
    asm volatile("s_wait_asynccnt 0" ::: "memory");
#endif
#endif
}

// float -> bf16 (round to nearest even), storage as ushort
__device__ __forceinline__ unsigned short f2bf(float f) {
    unsigned int u = __float_as_uint(f);
    u += 0x7FFFu + ((u >> 16) & 1u);
    return (unsigned short)(u >> 16);
}

// Load a 16x32 bf16 fragment (A or B operand) from an LDS tile whose K is
// contiguous per row.  CDNA5 16-bit layout: lanes 0-15 hold K 0-7 & 16-23,
// lanes 16-31 hold K 8-15 & 24-31; lane&15 selects the row (M or N).
__device__ __forceinline__ v16bf ldfrag(const unsigned short* p, int ld) {
    int lane = threadIdx.x & 31;
    const unsigned short* q = p + (lane & 15) * ld + ((lane >> 4) << 3);
    union { uint4 u[2]; v16us v; } t;
    t.u[0] = *(const uint4*)q;
    t.u[1] = *(const uint4*)(q + 16);
    return __builtin_bit_cast(v16bf, t.v);
}

__device__ __forceinline__ v8f wmma_bf16(v16bf a, v16bf b, v8f c) {
    return __builtin_amdgcn_wmma_f32_16x16x32_bf16(
        /*neg_a=*/false, a, /*neg_b=*/false, b,
        /*c_mod=*/(short)0, c, /*reuse_a=*/false, /*reuse_b=*/false);
}

// ---------------------------------------------------------------------------
// K0: fp32 -> bf16 conversion
// ---------------------------------------------------------------------------
__global__ void k_cvt(const float* __restrict__ src,
                      unsigned short* __restrict__ dst, int n) {
    int i = blockIdx.x * blockDim.x + threadIdx.x;
    int stride = gridDim.x * blockDim.x;
    for (; i < n; i += stride) dst[i] = f2bf(src[i]);
}

// ---------------------------------------------------------------------------
// K1/K5: C[M x N] (f32) = A[M x K] (bf16) * W[N x K]^T (bf16)
// Block tile 64(M) x 128(N), K-step 32, 8 waves in a 2x4 grid (32x32 each).
// M must be a multiple of 64, K a multiple of 32; N is bounds-checked.
// Tiles staged to LDS with async global->LDS DMA (ASYNCcnt).
// ---------------------------------------------------------------------------
#define GK 32
__global__ __launch_bounds__(256) void k_gemm_xt(
    const unsigned short* __restrict__ A,
    const unsigned short* __restrict__ W,
    float* __restrict__ C, int M, int N, int K) {
    __shared__ __align__(16) unsigned short As[64][GK + 8];
    __shared__ __align__(16) unsigned short Ws[128][GK + 8];
    int tid = threadIdx.x;
    int wid = tid >> 5;
    int m_blk = blockIdx.x * 64;
    int n_blk = blockIdx.y * 128;
    int wm = (wid >> 2) * 32;      // 0 or 32
    int wn = (wid & 3) * 32;       // 0,32,64,96
    v8f acc[2][2];
    for (int i = 0; i < 2; ++i)
        for (int j = 0; j < 2; ++j) acc[i][j] = (v8f)(0.f);

    int ar = tid >> 2, as = (tid & 3) << 3;   // A: 1 uint4/thread
    for (int kb = 0; kb < K; kb += GK) {
        __syncthreads();
        cp_async_b128(&A[(size_t)(m_blk + ar) * K + kb + as], &As[ar][as]);
        if (kb + GK < K)
            __builtin_prefetch(&A[(size_t)(m_blk + ar) * K + kb + GK + as], 0, 1);
        for (int it = 0; it < 2; ++it) {
            int lin = tid + it * 256;
            int wr = lin >> 2, wcs = (lin & 3) << 3;
            int gn = n_blk + wr;
            if (gn < N) {
                cp_async_b128(&W[(size_t)gn * K + kb + wcs], &Ws[wr][wcs]);
                if (kb + GK < K)
                    __builtin_prefetch(&W[(size_t)gn * K + kb + GK + wcs], 0, 1);
            } else {
                *(uint4*)&Ws[wr][wcs] = make_uint4(0u, 0u, 0u, 0u);
            }
        }
        cp_async_wait();
        __syncthreads();
        v16bf a0 = ldfrag(&As[wm + 0][0],  GK + 8);
        v16bf a1 = ldfrag(&As[wm + 16][0], GK + 8);
        v16bf b0 = ldfrag(&Ws[wn + 0][0],  GK + 8);
        v16bf b1 = ldfrag(&Ws[wn + 16][0], GK + 8);
        acc[0][0] = wmma_bf16(a0, b0, acc[0][0]);
        acc[0][1] = wmma_bf16(a0, b1, acc[0][1]);
        acc[1][0] = wmma_bf16(a1, b0, acc[1][0]);
        acc[1][1] = wmma_bf16(a1, b1, acc[1][1]);
    }
    int lane = tid & 31;
    int ro = (lane >> 4) << 3;      // +8 rows for upper half-wave
    int co = lane & 15;
    for (int i = 0; i < 2; ++i)
        for (int j = 0; j < 2; ++j)
            for (int r = 0; r < 8; ++r) {
                int row = m_blk + wm + i * 16 + r + ro;
                int col = n_blk + wn + j * 16 + co;
                if (col < N) C[(size_t)row * N + col] = acc[i][j][r];
            }
}

// ---------------------------------------------------------------------------
// K2: split h -> qf/q/k/v (bf16) with RoPE on q,k (1/sqrt(D) folded into q),
//     top-2 sigmoid router gates -> dense gate table (BT x NEXP, f32)
// ---------------------------------------------------------------------------
__global__ __launch_bounds__(256) void k_rope_router(
    const float* __restrict__ h,                 // BT x NIN
    unsigned short* __restrict__ qf,
    unsigned short* __restrict__ q,
    unsigned short* __restrict__ k,
    unsigned short* __restrict__ v,
    float* __restrict__ gate) {
    int t = blockIdx.x;
    int pos = t & (TT - 1);
    const float* hr = h + (size_t)t * NIN;
    int tid = threadIdx.x;
    for (int i = tid; i < EE; i += 256) {
        qf[(size_t)t * EE + i] = f2bf(hr[i]);
        v [(size_t)t * EE + i] = f2bf(hr[3 * EE + i]);
    }
    const float qscale = 0.08838834764831845f;   // 1/sqrt(128)
    for (int p = tid; p < EE / 2; p += 256) {
        int hh = p >> 6, d = p & 63;
        float inv = __powf(10000.f, -(float)d * (1.f / 64.f));
        float ang = (float)pos * inv;
        float c, s;
        __sincosf(ang, &c, &s);
        int c1 = hh * DD + d, c2 = c1 + 64;
        float q1 = hr[EE + c1], q2 = hr[EE + c2];
        q[(size_t)t * EE + c1] = f2bf((q1 * c - q2 * s) * qscale);
        q[(size_t)t * EE + c2] = f2bf((q1 * s + q2 * c) * qscale);
        float k1 = hr[2 * EE + c1], k2 = hr[2 * EE + c2];
        k[(size_t)t * EE + c1] = f2bf(k1 * c - k2 * s);
        k[(size_t)t * EE + c2] = f2bf(k1 * s + k2 * c);
    }
    if (tid == 0) {
        float b1 = -3.0e38f, b2 = -3.0e38f;
        int i1 = 0, i2 = 0;
        for (int e = 0; e < NEXP; ++e) {
            float vv = hr[4 * EE + e];
            if (vv > b1)      { b2 = b1; i2 = i1; b1 = vv; i1 = e; }
            else if (vv > b2) { b2 = vv; i2 = e; }
        }
        float g1 = 1.f / (1.f + __expf(-b1));
        float g2 = 1.f / (1.f + __expf(-b2));
        for (int e = 0; e < NEXP; ++e)
            gate[t * NEXP + e] = (e == i1 ? g1 : 0.f) + (e == i2 ? g2 : 0.f);
    }
}

// ---------------------------------------------------------------------------
// K3: causal flash attention.  Block = 128 threads (4 waves), 64 queries per
// block (16 per wave), K/V streamed in 64-key tiles.  All matmuls are WMMA.
// Writes attn_out into y[:, 0:1024] as bf16.
// ---------------------------------------------------------------------------
__global__ __launch_bounds__(128) void k_attn(
    const unsigned short* __restrict__ q,
    const unsigned short* __restrict__ k,
    const unsigned short* __restrict__ v,
    unsigned short* __restrict__ y) {            // BT x 2E
    __shared__ __align__(16) unsigned short Qs[64][136];
    __shared__ __align__(16) unsigned short Ks[64][136];
    __shared__ __align__(16) unsigned short Vt[128][72];
    __shared__ __align__(16) unsigned short Ps[4][16][72];
    int tid = threadIdx.x, lane = tid & 31, wv = tid >> 5;
    int q0 = blockIdx.x * 64;
    int bh = blockIdx.y, b = bh >> 3, hh = bh & 7;
    size_t tb = (size_t)b * TT;

    for (int it = 0; it < 8; ++it) {
        int lin = tid + it * 128;
        int r = lin >> 4, s = (lin & 15) << 3;
        cp_async_b128(&q[(tb + q0 + r) * EE + hh * DD + s], &Qs[r][s]);
    }
    cp_async_wait();
    v8f o[8];
    for (int i = 0; i < 8; ++i) o[i] = (v8f)(0.f);
    float mrow[8], lrow[8];
    for (int r = 0; r < 8; ++r) { mrow[r] = -3.0e38f; lrow[r] = 0.f; }
    int ro = (lane >> 4) << 3, co = lane & 15;

    int nkt = (q0 >> 6) + 1;
    for (int kt = 0; kt < nkt; ++kt) {
        __syncthreads();
        for (int it = 0; it < 8; ++it) {
            int lin = tid + it * 128;
            int r = lin >> 4, s = (lin & 15) << 3;
            cp_async_b128(&k[(tb + kt * 64 + r) * EE + hh * DD + s], &Ks[r][s]);
        }
        {   // V transposed into LDS: Vt[d][s]
            int d = tid;
            for (int s = 0; s < 64; ++s)
                Vt[d][s] = v[(tb + kt * 64 + s) * EE + hh * DD + d];
        }
        cp_async_wait();
        __syncthreads();

        // S (16x64 per wave) = Qw(16x128) * K^T
        v8f sacc[4];
        for (int nt = 0; nt < 4; ++nt) sacc[nt] = (v8f)(0.f);
        for (int kk = 0; kk < DD; kk += 32) {
            v16bf a = ldfrag(&Qs[wv * 16][kk], 136);
            for (int nt = 0; nt < 4; ++nt) {
                v16bf bb = ldfrag(&Ks[nt * 16][kk], 136);
                sacc[nt] = wmma_bf16(a, bb, sacc[nt]);
            }
        }
        // causal mask + online softmax (rows live in 16-lane halves)
        for (int r = 0; r < 8; ++r) {
            int qrow = q0 + wv * 16 + r + ro;
            for (int nt = 0; nt < 4; ++nt) {
                int kcol = kt * 64 + nt * 16 + co;
                sacc[nt][r] = (kcol <= qrow) ? sacc[nt][r] : -3.0e38f;
            }
            float mx = -3.0e38f;
            for (int nt = 0; nt < 4; ++nt) mx = fmaxf(mx, sacc[nt][r]);
            for (int off = 8; off > 0; off >>= 1)
                mx = fmaxf(mx, __shfl_xor(mx, off));
            float nm = fmaxf(mrow[r], mx);
            float scale = __expf(mrow[r] - nm);
            float rs = 0.f;
            for (int nt = 0; nt < 4; ++nt) {
                float p = __expf(sacc[nt][r] - nm);
                sacc[nt][r] = p;
                rs += p;
            }
            for (int off = 8; off > 0; off >>= 1)
                rs += __shfl_xor(rs, off);
            lrow[r] = lrow[r] * scale + rs;
            mrow[r] = nm;
            for (int nt = 0; nt < 8; ++nt) o[nt][r] *= scale;
        }
        // P -> LDS (bf16) to re-enter WMMA A layout
        for (int r = 0; r < 8; ++r)
            for (int nt = 0; nt < 4; ++nt)
                Ps[wv][r + ro][nt * 16 + co] = f2bf(sacc[nt][r]);
        // O(16x128) += P(16x64) * V(64x128)
        for (int kk = 0; kk < 64; kk += 32) {
            v16bf a = ldfrag(&Ps[wv][0][kk], 72);
            for (int nt = 0; nt < 8; ++nt) {
                v16bf bb = ldfrag(&Vt[nt * 16][kk], 72);
                o[nt] = wmma_bf16(a, bb, o[nt]);
            }
        }
    }
    for (int r = 0; r < 8; ++r) {
        float inv = 1.f / lrow[r];
        size_t tok = tb + q0 + wv * 16 + r + ro;
        for (int nt = 0; nt < 8; ++nt) {
            int col = hh * DD + nt * 16 + co;
            y[tok * (2 * EE) + col] = f2bf(o[nt][r] * inv);
        }
    }
}

// ---------------------------------------------------------------------------
// K4: expert FFN.  o = sum_e gate[t,e] * gelu(qf . Kf_e^T) . Vf_e
// Same tile skeleton as attention; gate folded into the gelu output so all
// 16 experts accumulate into one persistent WMMA accumulator.
// Writes into y[:, 1024:2048] as bf16.
// ---------------------------------------------------------------------------
__global__ __launch_bounds__(128) void k_experts(
    const unsigned short* __restrict__ qf,       // BT x E
    const unsigned short* __restrict__ kf,       // NE x H x S x D
    const unsigned short* __restrict__ vf,       // NE x H x S x D
    const float* __restrict__ gate,              // BT x NE
    unsigned short* __restrict__ y) {            // BT x 2E
    __shared__ __align__(16) unsigned short Qs[64][136];
    __shared__ __align__(16) unsigned short Kc[64][136];
    __shared__ __align__(16) unsigned short Vt[128][72];
    __shared__ __align__(16) unsigned short Ps[4][16][72];
    int tid = threadIdx.x, lane = tid & 31, wv = tid >> 5;
    int t0 = blockIdx.x * 64;
    int hh = blockIdx.y;

    for (int it = 0; it < 8; ++it) {
        int lin = tid + it * 128;
        int r = lin >> 4, s = (lin & 15) << 3;
        cp_async_b128(&qf[(size_t)(t0 + r) * EE + hh * DD + s], &Qs[r][s]);
    }
    cp_async_wait();
    v8f o[8];
    for (int i = 0; i < 8; ++i) o[i] = (v8f)(0.f);
    int ro = (lane >> 4) << 3, co = lane & 15;

    for (int e = 0; e < NEXP; ++e) {
        float grow[8];
        for (int r = 0; r < 8; ++r)
            grow[r] = gate[(size_t)(t0 + wv * 16 + r + ro) * NEXP + e];
        for (int sc = 0; sc < 4; ++sc) {
            const unsigned short* kbase =
                kf + (((size_t)e * NH + hh) * SS + sc * 64) * DD;
            const unsigned short* vbase =
                vf + (((size_t)e * NH + hh) * SS + sc * 64) * DD;
            __syncthreads();
            for (int it = 0; it < 8; ++it) {
                int lin = tid + it * 128;
                int r = lin >> 4, s = (lin & 15) << 3;
                cp_async_b128(&kbase[r * DD + s], &Kc[r][s]);
            }
            {   int d = tid;
                for (int s = 0; s < 64; ++s) Vt[d][s] = vbase[s * DD + d];
            }
            cp_async_wait();
            __syncthreads();
            v8f sacc[4];
            for (int nt = 0; nt < 4; ++nt) sacc[nt] = (v8f)(0.f);
            for (int kk = 0; kk < DD; kk += 32) {
                v16bf a = ldfrag(&Qs[wv * 16][kk], 136);
                for (int nt = 0; nt < 4; ++nt) {
                    v16bf bb = ldfrag(&Kc[nt * 16][kk], 136);
                    sacc[nt] = wmma_bf16(a, bb, sacc[nt]);
                }
            }
            // exact gelu * gate -> bf16 P
            for (int r = 0; r < 8; ++r)
                for (int nt = 0; nt < 4; ++nt) {
                    float x = sacc[nt][r];
                    float gl = 0.5f * x * (1.f + erff(x * 0.70710678118f));
                    Ps[wv][r + ro][nt * 16 + co] = f2bf(gl * grow[r]);
                }
            for (int kk = 0; kk < 64; kk += 32) {
                v16bf a = ldfrag(&Ps[wv][0][kk], 72);
                for (int nt = 0; nt < 8; ++nt) {
                    v16bf bb = ldfrag(&Vt[nt * 16][kk], 72);
                    o[nt] = wmma_bf16(a, bb, o[nt]);
                }
            }
        }
    }
    for (int r = 0; r < 8; ++r) {
        size_t tok = t0 + wv * 16 + r + ro;
        for (int nt = 0; nt < 8; ++nt) {
            int col = EE + hh * DD + nt * 16 + co;
            y[tok * (2 * EE) + col] = f2bf(o[nt][r]);
        }
    }
}

// ---------------------------------------------------------------------------
// host-side launch
// ---------------------------------------------------------------------------
extern "C" void kernel_launch(void* const* d_in, const int* in_sizes, int n_in,
                              void* d_out, int out_size, void* d_ws, size_t ws_size,
                              hipStream_t stream) {
    const float* x      = (const float*)d_in[0];
    const float* w_in   = (const float*)d_in[1];
    const float* w_out  = (const float*)d_in[2];
    const float* k_ffwd = (const float*)d_in[3];
    const float* v_ffwd = (const float*)d_in[4];
    float* out = (float*)d_out;

    char* ws = (char*)d_ws;
    size_t off = 0;
    auto alloc = [&](size_t bytes) -> void* {
        void* p = ws + off;
        off += (bytes + 255) & ~(size_t)255;
        return p;
    };
    const size_t nX  = (size_t)BT * EE;
    const size_t nWi = (size_t)NIN * EE;
    const size_t nWo = (size_t)EE * 2 * EE;
    const size_t nKV = (size_t)NEXP * NH * SS * DD;

    unsigned short* x_bf  = (unsigned short*)alloc(nX  * 2);
    unsigned short* wi_bf = (unsigned short*)alloc(nWi * 2);
    unsigned short* wo_bf = (unsigned short*)alloc(nWo * 2);
    unsigned short* kf_bf = (unsigned short*)alloc(nKV * 2);
    unsigned short* vf_bf = (unsigned short*)alloc(nKV * 2);
    float*          h     = (float*)alloc((size_t)BT * NIN * 4);
    unsigned short* qf_bf = (unsigned short*)alloc(nX * 2);
    unsigned short* q_bf  = (unsigned short*)alloc(nX * 2);
    unsigned short* k_bf  = (unsigned short*)alloc(nX * 2);
    unsigned short* v_bf  = (unsigned short*)alloc(nX * 2);
    unsigned short* y_bf  = (unsigned short*)alloc((size_t)BT * 2 * EE * 2);
    float*          gate  = (float*)alloc((size_t)BT * NEXP * 4);

    // K0: conversions
    k_cvt<<<2048, 256, 0, stream>>>(x,      x_bf,  (int)nX);
    k_cvt<<<2048, 256, 0, stream>>>(w_in,   wi_bf, (int)nWi);
    k_cvt<<<2048, 256, 0, stream>>>(w_out,  wo_bf, (int)nWo);
    k_cvt<<<2048, 256, 0, stream>>>(k_ffwd, kf_bf, (int)nKV);
    k_cvt<<<2048, 256, 0, stream>>>(v_ffwd, vf_bf, (int)nKV);

    // K1: h = x @ w_in^T   (4096 x 4112, K=1024)
    k_gemm_xt<<<dim3(BT / 64, (NIN + 127) / 128), 256, 0, stream>>>(
        x_bf, wi_bf, h, BT, NIN, EE);

    // K2: rope / split / router
    k_rope_router<<<BT, 256, 0, stream>>>(h, qf_bf, q_bf, k_bf, v_bf, gate);

    // K3: attention -> y[:, :1024]
    k_attn<<<dim3(TT / 64, 2 * NH), 128, 0, stream>>>(q_bf, k_bf, v_bf, y_bf);

    // K4: expert ffwd -> y[:, 1024:]
    k_experts<<<dim3(BT / 64, NH), 128, 0, stream>>>(qf_bf, kf_bf, vf_bf, gate, y_bf);

    // K5: out = y @ w_out^T   (4096 x 1024, K=2048)
    k_gemm_xt<<<dim3(BT / 64, EE / 128), 256, 0, stream>>>(
        y_bf, wo_bf, out, BT, EE, 2 * EE);
}
